// CrossWinAttention_72524817760870
// MI455X (gfx1250) — matrix operand order repaired and checked by reading, not compiled, not run on hardware
//
#include <hip/hip_runtime.h>
#include <hip/hip_bf16.h>
#include <cstdint>

typedef __attribute__((ext_vector_type(16))) _Float16 v16h;
typedef __attribute__((ext_vector_type(8)))  _Float16 v8h;
typedef __attribute__((ext_vector_type(8)))  float    v8f;

#define BB   2
#define NN   6
#define XX   8
#define YY   8
#define W1C  8
#define W2C  8
#define DD   128
#define HH   4
#define DHC  32
#define LL   64     // X*Y
#define QTC  384    // N*W1*W2
#define EPSC 1e-5f

#define WLD  (DD + 16)   // padded row stride (f16) for transposed weights in LDS
#define LNLD (DD + 8)    // padded row stride (f16) for activation tiles in LDS
#define SLD  (QTC + 1)   // padded row stride (f32) for scores
#define PLD  (QTC + 8)   // padded row stride (f16) for probs

// A-fragment (16x32 f16, MxK) from a row-major tile.  Lane layout:
// m = lane&15, hi = lane>>4;  K = hi*16+e (contiguous, matches B).
// One contiguous run -> two 16-byte loads.
static __device__ inline v16h frag_a_rowmajor(const _Float16* base, int ld, int lane) {
  int m = lane & 15, hi = lane >> 4;
  const _Float16* row = base + (size_t)m * ld + hi * 16;
  v8h a0 = *(const v8h*)(row);
  v8h a1 = *(const v8h*)(row + 8);
  return __builtin_shufflevector(a0, a1, 0, 1, 2, 3, 4, 5, 6, 7,
                                 8, 9, 10, 11, 12, 13, 14, 15);
}

// B-fragment (32x16 f16, KxN), K-innermost storage for fixed column:
// lane holds n = lane&15, k = (lane>>4)*16 + e -> 16 contiguous f16 = 32 bytes.
static __device__ inline v16h frag_b_kcontig(const _Float16* p) {
  return *(const v16h*)p;
}

// ---------------------------------------------------------------------------
// Kernel 1: window gather + LayerNorm + GEMM (D x D) + bias -> f16.
// TRANSV == false : dst layout [b][l][h][t][dh]   (Q, K)
// TRANSV == true  : dst layout [b][l][h][dh][t]   (V, so PV B-frags are contiguous)
// grid = (B*L, QT/64), block = 128 (4 waves).
// ---------------------------------------------------------------------------
template <bool TRANSV>
__global__ __launch_bounds__(128) void k_ln_qkv(
    const float* __restrict__ src, const float* __restrict__ gamma,
    const float* __restrict__ beta, const float* __restrict__ W,
    const float* __restrict__ bias, _Float16* __restrict__ dst) {
  __shared__ __align__(32) _Float16 Wl[DD * WLD];    // transposed: [n][k]
  __shared__ __align__(32) _Float16 lnm[64 * LNLD];  // [token][d]

  int tid = threadIdx.x;
  int win = blockIdx.x;   // 0..B*L-1
  int tt  = blockIdx.y;   // 0..5  (token tile == view index n)
  int b = win / LL, l = win % LL;
  int x = l / YY, y = l % YY;

  // coalesced read of W, transposed store to LDS
  for (int i = tid; i < DD * DD; i += 128) {
    int k = i >> 7, n = i & 127;
    Wl[n * WLD + k] = (_Float16)W[i];
  }

  if (tid < 64) {
    int r = tid;
    int w1 = r >> 3, w2 = r & 7;
    size_t gbase = ((((((size_t)b * NN + tt) * XX + x) * YY + y) * W1C + w1) * W2C + w2) * DD;
    const float4* row4 = (const float4*)(src + gbase);
    float s = 0.f, s2 = 0.f;
#pragma unroll 8
    for (int c = 0; c < DD / 4; ++c) {
      float4 vv = row4[c];
      s  += vv.x + vv.y + vv.z + vv.w;
      s2 += vv.x * vv.x + vv.y * vv.y + vv.z * vv.z + vv.w * vv.w;
    }
    float mean = s * (1.f / DD);
    float var  = s2 * (1.f / DD) - mean * mean;
    float inv  = rsqrtf(var + EPSC);
#pragma unroll 8
    for (int c = 0; c < DD / 4; ++c) {
      float4 vv = row4[c];
      const float4 g4 = ((const float4*)gamma)[c];
      const float4 b4 = ((const float4*)beta)[c];
      lnm[r * LNLD + 4 * c + 0] = (_Float16)((vv.x - mean) * inv * g4.x + b4.x);
      lnm[r * LNLD + 4 * c + 1] = (_Float16)((vv.y - mean) * inv * g4.y + b4.y);
      lnm[r * LNLD + 4 * c + 2] = (_Float16)((vv.z - mean) * inv * g4.z + b4.z);
      lnm[r * LNLD + 4 * c + 3] = (_Float16)((vv.w - mean) * inv * g4.w + b4.w);
    }
  }
  __syncthreads();

  int wave = tid >> 5, lane = tid & 31;
  int hi = lane >> 4, lo = lane & 15;
  int m0 = wave * 16;
  for (int nt = 0; nt < 8; ++nt) {
    v8f c = {};
#pragma unroll
    for (int kc = 0; kc < 4; ++kc) {
      v16h a  = frag_a_rowmajor(&lnm[m0 * LNLD + kc * 32], LNLD, lane);
      v16h bm = frag_b_kcontig(&Wl[(nt * 16 + lo) * WLD + kc * 32 + hi * 16]);
      c = __builtin_amdgcn_wmma_f32_16x16x32_f16(false, a, false, bm, (short)0, c, false, false);
    }
    int col = nt * 16 + lo;
    float bb = bias[col];
    int h = col >> 5, dh = col & 31;
    if (TRANSV) {
      v8h ov;
#pragma unroll
      for (int r = 0; r < 8; ++r) ov[r] = (_Float16)(c[r] + bb);
      size_t t0 = (size_t)tt * 64 + m0 + hi * 8;   // 8 consecutive tokens
      size_t o = ((((size_t)(b * LL + l)) * HH + h) * DHC + dh) * QTC + t0;
      *(v8h*)&dst[o] = ov;
    } else {
#pragma unroll
      for (int r = 0; r < 8; ++r) {
        int m = m0 + r + hi * 8;
        int t = tt * 64 + m;
        size_t o = ((((size_t)(b * LL + l)) * HH + h) * QTC + t) * DHC + dh;
        dst[o] = (_Float16)(c[r] + bb);
      }
    }
  }
}

// ---------------------------------------------------------------------------
// Kernel 2: attention for one (b,l,h, 16-query tile).  One wave per block.
// Q,K layout [t][dh]; V layout [dh][t] (transposed by kernel 1).
// grid = B*L*H*(QT/16), block = 32.
// ---------------------------------------------------------------------------
__global__ __launch_bounds__(32) void k_attn(
    const _Float16* __restrict__ Q, const _Float16* __restrict__ K,
    const _Float16* __restrict__ V, _Float16* __restrict__ Aout) {
  __shared__ __align__(32) float    S[16 * SLD];   // ~24.6 KB
  __shared__ __align__(32) _Float16 P[16 * PLD];   // ~12.5 KB
  __shared__ float invs[16];

  int lane = threadIdx.x;
  int hi = lane >> 4, lo = lane & 15;
  int blk = blockIdx.x;
  int qt = blk % 24; blk /= 24;
  int h  = blk % HH; blk /= HH;
  int l  = blk % LL;
  int b  = blk / LL;

  size_t base = (((size_t)(b * LL + l)) * HH + h) * (size_t)QTC * DHC;
  const _Float16* Qp = Q + base;
  const _Float16* Kp = K + base;   // [t][dh]
  const _Float16* Vt = V + base;   // [dh][t]

  __builtin_prefetch((const void*)Kp, 0, 1);
  __builtin_prefetch((const void*)Vt, 0, 1);

  // Q A-fragment (K = DH = 32 -> single chunk), held in registers
  v16h aq;
  {
    int m = qt * 16 + lo;
    const _Float16* row = Qp + (size_t)m * DHC + hi * 16;
    v8h a0 = *(const v8h*)(row);
    v8h a1 = *(const v8h*)(row + 8);
    aq = __builtin_shufflevector(a0, a1, 0, 1, 2, 3, 4, 5, 6, 7,
                                 8, 9, 10, 11, 12, 13, 14, 15);
  }
  const float scale = 0.17677669529663687f;  // 1/sqrt(32)

  for (int kt = 0; kt < 24; ++kt) {
    // B = K^T (dh x keys): lane n = key = kt*16+lo, contiguous k = dh
    v16h bk = frag_b_kcontig(Kp + (size_t)(kt * 16 + lo) * DHC + hi * 16);
    v8f c = {};
    c = __builtin_amdgcn_wmma_f32_16x16x32_f16(false, aq, false, bk, (short)0, c, false, false);
#pragma unroll
    for (int r = 0; r < 8; ++r) {
      int m = r + hi * 8;
      S[m * SLD + kt * 16 + lo] = c[r] * scale;
    }
  }
  __syncthreads();

  if (lane < 16) {
    int r = lane;
    float mx = -3.0e38f;
    for (int j = 0; j < QTC; ++j) mx = fmaxf(mx, S[r * SLD + j]);
    float s = 0.f;
    for (int j = 0; j < QTC; ++j) {
      float e = __expf(S[r * SLD + j] - mx);
      P[r * PLD + j] = (_Float16)e;   // unnormalized probs
      s += e;
    }
    invs[r] = 1.f / s;
  }
  __syncthreads();

  for (int nt = 0; nt < 2; ++nt) {
    v8f c = {};
    for (int kc = 0; kc < 12; ++kc) {
      v16h ap = frag_a_rowmajor(&P[kc * 32], PLD, lane);
      // B = V chunk (keys x dh): lane n = dh = nt*16+lo, contiguous k = key
      v16h bv = frag_b_kcontig(Vt + (size_t)(nt * 16 + lo) * QTC + kc * 32 + hi * 16);
      c = __builtin_amdgcn_wmma_f32_16x16x32_f16(false, ap, false, bv, (short)0, c, false, false);
    }
#pragma unroll
    for (int r = 0; r < 8; ++r) {
      int m = r + hi * 8;
      int t = qt * 16 + m;
      int col = h * DHC + nt * 16 + lo;
      size_t o = (((size_t)(b * LL + l)) * QTC + t) * DD + col;
      Aout[o] = (_Float16)(c[r] * invs[m]);
    }
  }
}

// ---------------------------------------------------------------------------
// Kernel 3: mean over N views (folded before projection), GEMM with wp,
// + bias + skip -> fp32 output.  One block per window (128 threads, 4 waves).
// grid = B*L, block = 128.
// ---------------------------------------------------------------------------
__global__ __launch_bounds__(128) void k_proj(
    const _Float16* __restrict__ Aout, const float* __restrict__ Wp,
    const float* __restrict__ bp, const float* __restrict__ skip,
    float* __restrict__ out) {
  __shared__ __align__(32) _Float16 Wl[DD * WLD];    // transposed wp: [n][k]
  __shared__ __align__(32) _Float16 abar[64 * LNLD]; // averaged activations

  int tid = threadIdx.x;
  int blk = blockIdx.x;        // b*L + l
  int l = blk % LL;
  int b = blk / LL;
  int x = l / YY, y = l % YY;

  for (int i = tid; i < DD * DD; i += 128) {
    int k = i >> 7, n = i & 127;
    Wl[n * WLD + k] = (_Float16)Wp[i];
  }

  const _Float16* Ab = Aout + (((size_t)(b * LL + l)) * QTC) * DD;
  for (int idx = tid; idx < 64 * DD; idx += 128) {
    int row = idx >> 7, col = idx & 127;   // row == w1w2
    float s = 0.f;
#pragma unroll
    for (int n = 0; n < NN; ++n)
      s += (float)Ab[(size_t)(n * 64 + row) * DD + col];
    abar[row * LNLD + col] = (_Float16)(s * (1.f / NN));
  }
  __syncthreads();

  int wave = tid >> 5, lane = tid & 31;
  int hi = lane >> 4, lo = lane & 15;
  int m0 = wave * 16;
  for (int nt = 0; nt < 8; ++nt) {
    v8f c = {};
#pragma unroll
    for (int kc = 0; kc < 4; ++kc) {
      v16h a  = frag_a_rowmajor(&abar[m0 * LNLD + kc * 32], LNLD, lane);
      v16h bm = frag_b_kcontig(&Wl[(nt * 16 + lo) * WLD + kc * 32 + hi * 16]);
      c = __builtin_amdgcn_wmma_f32_16x16x32_f16(false, a, false, bm, (short)0, c, false, false);
    }
    int col = nt * 16 + lo;
    float bb = bp[col];
#pragma unroll
    for (int r = 0; r < 8; ++r) {
      int m = m0 + r + hi * 8;   // w1w2
      int w1 = m >> 3, w2 = m & 7;
      size_t o = (((((size_t)b * XX + x) * YY + y) * W1C + w1) * W2C + w2) * DD + col;
      out[o] = c[r] + bb + skip[o];
    }
  }
}

// ---------------------------------------------------------------------------
extern "C" void kernel_launch(void* const* d_in, const int* in_sizes, int n_in,
                              void* d_out, int out_size, void* d_ws, size_t ws_size,
                              hipStream_t stream) {
  const float* q    = (const float*)d_in[0];
  const float* k    = (const float*)d_in[1];
  const float* v    = (const float*)d_in[2];
  const float* skip = (const float*)d_in[3];
  const float* gq   = (const float*)d_in[4];
  const float* betq = (const float*)d_in[5];
  const float* gk   = (const float*)d_in[6];
  const float* betk = (const float*)d_in[7];
  const float* gv   = (const float*)d_in[8];
  const float* betv = (const float*)d_in[9];
  const float* wq   = (const float*)d_in[10];
  const float* bq   = (const float*)d_in[11];
  const float* wk   = (const float*)d_in[12];
  const float* bk   = (const float*)d_in[13];
  const float* wv   = (const float*)d_in[14];
  const float* bv   = (const float*)d_in[15];
  const float* wp   = (const float*)d_in[16];
  const float* bp   = (const float*)d_in[17];
  float* out = (float*)d_out;

  // Workspace: qh, kh ([b][l][h][t][dh]), vh ([b][l][h][dh][t]), ao ([b][l][t][hd]); all f16
  size_t per = (size_t)BB * LL * QTC * DD;  // 6,291,456 elements each
  _Float16* qh = (_Float16*)d_ws;
  _Float16* kh = qh + per;
  _Float16* vh = kh + per;
  _Float16* ao = vh + per;

  dim3 g1(BB * LL, QTC / 64);
  k_ln_qkv<false><<<g1, 128, 0, stream>>>(q, gq, betq, wq, bq, qh);
  k_ln_qkv<false><<<g1, 128, 0, stream>>>(k, gk, betk, wk, bk, kh);
  k_ln_qkv<true><<<g1, 128, 0, stream>>>(v, gv, betv, wv, bv, vh);

  k_attn<<<BB * LL * HH * (QTC / 16), 32, 0, stream>>>(qh, kh, vh, ao);

  k_proj<<<BB * LL, 128, 0, stream>>>(ao, wp, bp, skip, out);
}
